// WENONetwork_44770739093965
// MI455X (gfx1250) — matrix operand order repaired and checked by reading, not compiled, not run on hardware
//
#include <hip/hip_runtime.h>

typedef _Float16 half_t;
typedef __attribute__((ext_vector_type(16))) _Float16 v16h;
typedef __attribute__((ext_vector_type(8)))  _Float16 v8h;
typedef __attribute__((ext_vector_type(8)))  float    v8f;

#define WAVES     8
#define BLOCK_T   (WAVES * 32)     // 256 threads = 8 wave32
#define MT        256              // mult positions per block (16 col-tiles)
#define OUT_TILE  252              // outputs per block (~98% lane util in stage E)
#define C1_TILES  17               // conv1 column tiles (272 cols)
#define SU_LEN    272              // staged uu: positions [m0-6, m0+MT+6]
#define D_LEN     272
#define D_VALID   267              // d1/d2 valid: positions [m0-5, m0+MT+5]
#define H1_COLS   272

__device__ __forceinline__ float eluf(float x) {
  return x > 0.f ? x : (__expf(x) - 1.f);
}
__device__ __forceinline__ float sigmoidf(float x) {
  return 1.f / (1.f + __expf(-x));
}
// lane <-> lane^16 exchange (group-of-32 ds_swizzle SWAPX16: and=0x1f,or=0,xor=0x10)
__device__ __forceinline__ float xor16f(float x) {
  return __int_as_float(__builtin_amdgcn_ds_swizzle(__float_as_int(x), 0x401F));
}

// Mapped (Henrick) mweno WENO5 one-sided reconstruction, f32.
__device__ __forceinline__ float weno_side(float W0, float W1, float W2,
                                           float W3, float W4,
                                           float M0, float M1, float M2) {
  const float EPS = 1e-13f;
  const float g0 = 0.1f, g1 = 0.6f, g2 = 0.3f;
  float f0 = (11.f*W2 - 7.f*W3 + 2.f*W4) * (1.f/6.f);
  float f1 = ( 2.f*W1 + 5.f*W2 -     W3) * (1.f/6.f);
  float f2 = (   -W0  + 5.f*W1 + 2.f*W2) * (1.f/6.f);
  float t;
  t = W2 - 2.f*W3 + W4;       float ob0 = (13.f/12.f)*t*t;
  t = 3.f*W2 - 4.f*W3 + W4;   ob0 += 0.25f*t*t;
  t = W1 - 2.f*W2 + W3;       float ob1 = (13.f/12.f)*t*t;
  t = W1 - W3;                ob1 += 0.25f*t*t;
  t = W0 - 2.f*W1 + W2;       float ob2 = (13.f/12.f)*t*t;
  t = W0 - 4.f*W1 + 3.f*W2;   ob2 += 0.25f*t*t;
  float b0 = ob0 * M2, b1 = ob1 * M1, b2 = ob2 * M0;   // mult_shifts [1,0,-1]
  float brs = ob2 - ob0; brs *= brs;
  float e0 = (EPS + b0) * (EPS + b0);
  float e1 = (EPS + b1) * (EPS + b1);
  float e2 = (EPS + b2) * (EPS + b2);
  float om0 = g0 / e0 * (brs + e0);
  float om1 = g1 / e1 * (brs + e1);
  float om2 = g2 / e2 * (brs + e2);
  float inv = 1.f / (om0 + om1 + om2);
  om0 *= inv; om1 *= inv; om2 *= inv;
  float a0 = om0*(g0 + g0*g0 - 3.f*g0*om0 + om0*om0) / (g0*g0 + om0*(1.f - 2.f*g0));
  float a1 = om1*(g1 + g1*g1 - 3.f*g1*om1 + om1*om1) / (g1*g1 + om1*(1.f - 2.f*g1));
  float a2 = om2*(g2 + g2*g2 - 3.f*g2*om2 + om2*om2) / (g2*g2 + om2*(1.f - 2.f*g2));
  float inv2 = 1.f / (a0 + a1 + a2);
  return (a0*f0 + a1*f1 + a2*f2) * inv2;
}

// A-matrix (16x32 f16) element->K map per ISA 7.12.2:
// lanes 0-15 : e<8 -> K=e   ; e>=8 -> K=e+8
// lanes 16-31: e<8 -> K=e+8 ; e>=8 -> K=e+16
__device__ __forceinline__ int a_kmap(int e, bool hi) {
  return e + (e < 8 ? (hi ? 8 : 0) : (hi ? 16 : 8));
}

__global__ __launch_bounds__(BLOCK_T)
void weno_nn_kernel(const float* __restrict__ uu,
                    const float* __restrict__ w1, const float* __restrict__ b1,
                    const float* __restrict__ w2, const float* __restrict__ b2,
                    const float* __restrict__ w3, const float* __restrict__ b3,
                    float* __restrict__ out, int N, int NOUT)
{
  __shared__ __align__(32) float  su[SU_LEN];
  __shared__ __align__(32) half_t sd1[D_LEN];
  __shared__ __align__(32) half_t sd2[D_LEN];
  __shared__ __align__(32) half_t sb1t[H1_COLS * 16];   // conv1 B panel: [col][k0..15]
  __shared__ __align__(32) half_t sh1t[H1_COLS * 32];   // h1 transposed: [col][ch0..31]
  __shared__ __align__(32) half_t sw2a[10 * 32 * 16];   // conv2 A frags [tap*2+tile][lane][e]
  __shared__ __align__(32) float  smult[MT];

  const int  tid  = threadIdx.x;
  const int  lane = tid & 31;
  const int  wv   = tid >> 5;
  const int  n    = lane & 15;
  const bool hi   = lane >= 16;

  const int j0 = blockIdx.x * OUT_TILE;  // first output index of this block
  const int m0 = j0 + 2;                 // first mult position
  const int u0 = m0 - 6;                 // global position of su[0]

  // ---------- stage A: stage uu tile + halo (clamped) ----------
  for (int i = tid; i < SU_LEN; i += BLOCK_T) {
    int g = u0 + i;
    g = g < 0 ? 0 : (g > N - 1 ? N - 1 : g);
    su[i] = uu[g];
  }

  // ---------- stage A2: pre-swizzle conv2 A fragments into LDS (branch-free) ----
  // layout: frag f = tap*2 + tile ; per lane 16 contiguous halves (one v16h)
  for (int idx = tid; idx < 10 * 32 * 16; idx += BLOCK_T) {
    int f  = idx >> 9;            // /512
    int r  = idx & 511;
    int ln = r >> 4;
    int e  = r & 15;
    int tap  = f >> 1;
    int tile = f & 1;
    bool lhi = ln >= 16;
    int  lm  = ln & 15;
    int  K   = e + (lhi ? 16 : 0);    // input channel
    int  ch  = tile * 16 + lm;    // output channel
    bool valid = (K < 20) && (ch < 20);
    int  Kc  = K  < 20 ? K  : 19;
    int  chc = ch < 20 ? ch : 19;
    float v = w2[(chc * 20 + Kc) * 5 + tap];   // unconditional load, clamped index
    sw2a[idx] = (half_t)(valid ? v : 0.f);
  }
  __syncthreads();

  // ---------- stage B: averaged 1st/2nd differences -> f16 ----------
  for (int i = tid; i < D_VALID; i += BLOCK_T) {
    int p  = m0 - 5 + i;
    int pl = (p - 1 < 0) ? 0 : p - 1;
    int pr = (p < N - 2) ? p : N - 2;
    float dfl = su[pl + 1 - u0] - su[pl - u0];
    float dfr = su[pr + 1 - u0] - su[pr - u0];
    float d1v = 0.5f * (dfl + dfr);
    int pc = p < 1 ? 1 : (p > N - 2 ? N - 2 : p);
    float d2v = su[pc + 1 - u0] - 2.f * su[pc - u0] + su[pc - 1 - u0];
    sd1[i] = (half_t)d1v;
    sd2[i] = (half_t)d2v;
  }
  __syncthreads();

  // ---------- stage B2: pack conv1 B panel (im2col, K=10 padded to 16) ----------
  for (int i = tid; i < H1_COLS; i += BLOCK_T) {
    half_t* row = &sb1t[i * 16];
    #pragma unroll
    for (int tap = 0; tap < 5; ++tap) {
      int idx = i + tap + 1;                    // (q+tap-2) - (m0-5)
      idx = idx > D_VALID - 1 ? D_VALID - 1 : idx;
      row[tap]     = sd1[idx];
      row[5 + tap] = sd2[idx];
    }
    #pragma unroll
    for (int k = 10; k < 16; ++k) row[k] = (half_t)0.f;
  }

  // conv1 weight A fragments (branch-free clamped gather)
  v16h a1t0, a1t1;
  #pragma unroll
  for (int e = 0; e < 16; ++e) {
    int  k     = e + (hi ? 16 : 0);
    bool valid = k < 10;
    int  kc    = valid ? k : 0;
    int  ci    = kc >= 5;
    int  tap   = kc - ci * 5;
    int  ch1c  = (n < 4) ? (16 + n) : 19;
    float v0 = w1[(n * 2 + ci) * 5 + tap];
    float v1 = w1[(ch1c * 2 + ci) * 5 + tap];
    a1t0[e] = (half_t)(valid ? v0 : 0.f);
    a1t1[e] = (half_t)((valid && (n < 4)) ? v1 : 0.f);
  }
  __syncthreads();

  // ---------- stage C: conv1 via WMMA, write h1 transposed (zero-padded to 32 ch) ----
  for (int ct = wv; ct < C1_TILES; ct += WAVES) {
    const v16h zero16 = {};
    v16h bfrag = hi ? zero16 : *(const v16h*)&sb1t[(ct * 16 + n) * 16];
    v8f c0 = {0,0,0,0,0,0,0,0};
    v8f c1 = {0,0,0,0,0,0,0,0};
    c0 = __builtin_amdgcn_wmma_f32_16x16x32_f16(false, a1t0, false, bfrag,
                                                (short)0, c0, false, false);
    c1 = __builtin_amdgcn_wmma_f32_16x16x32_f16(false, a1t1, false, bfrag,
                                                (short)0, c1, false, false);
    int col = ct * 16 + n;
    // tile0: lanes<16 -> ch v, lanes>=16 -> ch 8+v (branch-free bias select)
    v8h p0, p1;
    #pragma unroll
    for (int v = 0; v < 8; ++v) {
      float blo = b1[v];          // uniform s_load, hoisted
      float bhi = b1[8 + v];      // uniform s_load, hoisted
      p0[v] = (half_t)eluf(c0[v] + (hi ? bhi : blo));
    }
    // tile1: lanes<16 -> ch 16+v (valid v<4), lanes>=16 -> ch 24+v (all pad)
    #pragma unroll
    for (int v = 0; v < 8; ++v) {
      float bch = (v < 4) ? b1[16 + v] : 0.f;   // compile-time index
      float x = eluf(c1[v] + bch);              // computed unconditionally
      p1[v] = (half_t)(((v < 4) && !hi) ? x : 0.f);
    }
    *(v8h*)&sh1t[col * 32 + (hi ? 8 : 0)]   = p0;
    *(v8h*)&sh1t[col * 32 + (hi ? 24 : 16)] = p1;
  }
  __syncthreads();

  // ---------- stage D: conv2 as 5 tap-GEMMs (K=20->32) + conv3/sigmoid ----------
  // Each wave handles 2 column sub-tiles: col-tile wv and wv+8.
  for (int st = 0; st < 2; ++st) {
    const int cb = (wv + st * WAVES) * 16;     // column base within MT
    v8f acc0 = {0,0,0,0,0,0,0,0};
    v8f acc1 = {0,0,0,0,0,0,0,0};
    #pragma unroll
    for (int tap = 0; tap < 5; ++tap) {
      v16h a0f = *(const v16h*)&sw2a[((tap * 2 + 0) * 32 + lane) * 16];
      v16h a1f = *(const v16h*)&sw2a[((tap * 2 + 1) * 32 + lane) * 16];
      // B: column m = m0 + cb + n ; h1t index = (m + tap - 2) - (m0 - 2)
      v16h bfrag = *(const v16h*)&sh1t[(cb + n + tap) * 32 + (hi ? 16 : 0)];
      acc0 = __builtin_amdgcn_wmma_f32_16x16x32_f16(false, a0f, false, bfrag,
                                                    (short)0, acc0, false, false);
      acc1 = __builtin_amdgcn_wmma_f32_16x16x32_f16(false, a1f, false, bfrag,
                                                    (short)0, acc1, false, false);
    }

    // conv3 (1x1, 20ch) + bias + sigmoid, branch-free from the D-fragments
    float part = 0.f;
    #pragma unroll
    for (int v = 0; v < 8; ++v) {
      float wlo = w3[v],     whi = w3[8 + v];   // uniform s_loads
      float blo = b2[v],     bhi = b2[8 + v];
      part += (hi ? whi : wlo) * eluf(acc0[v] + (hi ? bhi : blo));
    }
    #pragma unroll
    for (int v = 0; v < 4; ++v) {
      float x = w3[16 + v] * eluf(acc1[v] + b2[16 + v]);
      part += hi ? 0.f : x;
    }
    float tot = part + xor16f(part);   // combine lane l with lane l^16
    if (!hi) smult[cb + n] = sigmoidf(tot + b3[0]) + 0.1f;
  }
  __syncthreads();

  // ---------- stage E: WENO5 fluxes (f32) ----------
  if (tid < OUT_TILE) {
    int j = j0 + tid;
    if (j < NOUT) {
      float U0 = su[tid + 5],  U1 = su[tid + 6],  U2 = su[tid + 7];
      float U3 = su[tid + 8],  U4 = su[tid + 9],  U5 = su[tid + 10];
      float M0 = smult[tid], M1 = smult[tid + 1], M2 = smult[tid + 2];
      float fn = weno_side(U0, U1, U2, U3, U4, M0, M1, M2);
      float fp = weno_side(U1, U2, U3, U4, U5, M0, M1, M2);
      out[j] = fp - fn;
    }
  }
}

extern "C" void kernel_launch(void* const* d_in, const int* in_sizes, int n_in,
                              void* d_out, int out_size, void* d_ws, size_t ws_size,
                              hipStream_t stream) {
  (void)n_in; (void)d_ws; (void)ws_size;
  const float* uu = (const float*)d_in[0];
  const float* w1 = (const float*)d_in[1];
  const float* b1 = (const float*)d_in[2];
  const float* w2 = (const float*)d_in[3];
  const float* b2 = (const float*)d_in[4];
  const float* w3 = (const float*)d_in[5];
  const float* b3 = (const float*)d_in[6];
  int N    = in_sizes[0];
  int NOUT = out_size;                       // N - 6
  int blocks = (NOUT + OUT_TILE - 1) / OUT_TILE;
  weno_nn_kernel<<<blocks, BLOCK_T, 0, stream>>>(uu, w1, b1, w2, b2, w3, b3,
                                                 (float*)d_out, N, NOUT);
}
